// MultiHeadAttention_11596411699476
// MI455X (gfx1250) — matrix operand order repaired and checked by reading, not compiled, not run on hardware
//
#include <hip/hip_runtime.h>

#define B_   2
#define T_   2048
#define D_   1024
#define H_   16
#define DH_  64
#define EPS_ 1e-5f
#define SCALE_ 0.125f   // 1/sqrt(64)

typedef __attribute__((ext_vector_type(16))) __bf16 v16bf;
typedef __attribute__((ext_vector_type(8)))  float  v8f;

union FragBF { v16bf v; uint4 q[2]; };

static __device__ __forceinline__ unsigned short f2bf(float f) {
  unsigned u = __float_as_uint(f);
  unsigned r = u + 0x7FFFu + ((u >> 16) & 1u);   // round-to-nearest-even
  return (unsigned short)(r >> 16);
}

static __device__ __forceinline__ v8f zero8() {
  v8f z;
#pragma unroll
  for (int i = 0; i < 8; ++i) z[i] = 0.0f;
  return z;
}

// async 16-byte global -> LDS copy (per-lane), tracked by ASYNCcnt
static __device__ __forceinline__ void async_copy16(const void* lds_ptr, const void* gptr) {
  unsigned lds_off = (unsigned)(size_t)lds_ptr;                 // generic LDS addr[31:0] = LDS offset
  unsigned long long ga = (unsigned long long)(size_t)gptr;
  asm volatile("global_load_async_to_lds_b128 %0, %1, off"
               :: "v"(lds_off), "v"(ga) : "memory");
}
#define WAIT_ASYNC_LE(n) asm volatile("s_wait_asynccnt %0" :: "i"(n) : "memory")

// A-operand (16x32 bf16): lane m=L%16, lo=L/16 holds k in {8lo..8lo+7} u {16+8lo..16+8lo+7}
static __device__ __forceinline__ v16bf frag_a(const unsigned short* p, int ld, int lane) {
  const int m = lane & 15, lo = lane >> 4;
  FragBF f;
  f.q[0] = *(const uint4*)(p + m * ld + 8 * lo);
  f.q[1] = *(const uint4*)(p + m * ld + 16 + 8 * lo);
  return f.v;
}

// B-operand (32x16 bf16) from row-major [n][k] storage: lane n=L%16, hi=L/16 holds k in {8hi..8hi+7} u {16+8hi..16+8hi+7}
static __device__ __forceinline__ v16bf frag_b(const unsigned short* p, int ld, int lane) {
  const int n = lane & 15, hi = lane >> 4;
  const unsigned short* row = p + n * ld + 8 * hi;
  FragBF f;
  f.q[0] = *(const uint4*)(row);
  f.q[1] = *(const uint4*)(row + 16);
  return f.v;
}

#define WMMA_BF16(Am, Bm, Cm) \
  __builtin_amdgcn_wmma_f32_16x16x32_bf16(false, (Am), false, (Bm), (short)0, (Cm), false, false)

// ---------------------------------------------------------------- fp32 -> bf16
__global__ __launch_bounds__(256) void cvt_bf16_kernel(const float* __restrict__ in,
                                                       unsigned short* __restrict__ out, int n) {
  int i = blockIdx.x * 256 + threadIdx.x;
  if (i < n) out[i] = f2bf(in[i]);
}

// ---------------------------------------------------------------- QKV GEMM
// C[m][n] = X[m][:] . W[n][:] + b[n];  128x128 block, 8 waves (4x2), K-step 32
#define BM  128
#define BN  128
#define BK  32
#define LDT 40   // 32 + 8 pad (bf16 elems)

__global__ __launch_bounds__(256) void qkv_gemm_kernel(
    const float* __restrict__ qin, const float* __restrict__ kin, const float* __restrict__ vin,
    const unsigned short* __restrict__ wqkv,  // [3D, D] bf16
    const float* __restrict__ bqkv,           // [3D]
    unsigned short* __restrict__ oq, unsigned short* __restrict__ ok,
    unsigned short* __restrict__ ov) {
  __shared__ unsigned short As[BM * LDT];
  __shared__ unsigned short Bs[BN * LDT];

  const int tid = threadIdx.x, lane = tid & 31, wid = tid >> 5;
  const int wm = wid >> 1, wn = wid & 1;
  const int which = blockIdx.z;
  const float* X = (which == 0) ? qin : (which == 1) ? kin : vin;
  const unsigned short* W = wqkv + (size_t)which * D_ * D_;
  const float* bias = bqkv + which * D_;
  const int rowBase = blockIdx.y * BM;
  const int colBase = blockIdx.x * BN;

  v8f acc[2][4];
#pragma unroll
  for (int i = 0; i < 2; ++i)
#pragma unroll
    for (int j = 0; j < 4; ++j) acc[i][j] = zero8();

  for (int k0 = 0; k0 < D_; k0 += BK) {
    // stage B (bf16 weights): async direct-to-LDS, 2x16B per thread
#pragma unroll
    for (int it = 0; it < 2; ++it) {
      int li = it * 256 + tid;
      int row = li >> 2;
      int c8 = (li & 3) << 3;
      async_copy16(&Bs[row * LDT + c8], W + (size_t)(colBase + row) * D_ + k0 + c8);
    }
    // stage A: 128x32 fp32 -> bf16 via registers (each thread: 4 float4)
#pragma unroll
    for (int it = 0; it < 4; ++it) {
      int li = it * 256 + tid;
      int row = li >> 3;
      int c4 = (li & 7) << 2;
      float4 f = *(const float4*)(X + (size_t)(rowBase + row) * D_ + k0 + c4);
      uint2 u;
      u.x = (unsigned)f2bf(f.x) | ((unsigned)f2bf(f.y) << 16);
      u.y = (unsigned)f2bf(f.z) | ((unsigned)f2bf(f.w) << 16);
      *(uint2*)&As[row * LDT + c4] = u;
    }
    WAIT_ASYNC_LE(0);
    __syncthreads();

    v16bf a0 = frag_a(&As[(wm * 32) * LDT], LDT, lane);
    v16bf a1 = frag_a(&As[(wm * 32 + 16) * LDT], LDT, lane);
#pragma unroll
    for (int nt = 0; nt < 4; ++nt) {
      v16bf bb = frag_b(&Bs[(wn * 64 + nt * 16) * LDT], LDT, lane);
      acc[0][nt] = WMMA_BF16(a0, bb, acc[0][nt]);
      acc[1][nt] = WMMA_BF16(a1, bb, acc[1][nt]);
    }
    __syncthreads();
  }

  const int n16 = lane & 15, hi = lane >> 4;
#pragma unroll
  for (int mt = 0; mt < 2; ++mt)
#pragma unroll
    for (int nt = 0; nt < 4; ++nt)
#pragma unroll
      for (int r = 0; r < 8; ++r) {
        int gm = rowBase + wm * 32 + mt * 16 + r + 8 * hi;
        int gn = colBase + wn * 64 + nt * 16 + n16;
        float v = acc[mt][nt][r] + bias[gn];
        int b = gm >> 11, t = gm & (T_ - 1);
        int h = gn >> 6, dh = gn & 63;
        size_t bh = (size_t)(b * H_ + h);
        if (which == 0)
          oq[(bh * T_ + t) * DH_ + dh] = f2bf(v * SCALE_);          // Q pre-scaled
        else if (which == 1)
          ok[(bh * T_ + t) * DH_ + dh] = f2bf(v);                   // K row-major [t][dh]
        else
          ov[(bh * DH_ + dh) * T_ + t] = f2bf(v);                   // V transposed [dh][t]
      }
}

// ---------------------------------------------------------------- flash attention + LN
#define KT  64
#define LDK 72   // 64 + 8 pad
#define NKT (T_ / KT)

__global__ __launch_bounds__(128) void attn_kernel(
    const unsigned short* __restrict__ wq, const unsigned short* __restrict__ wk,
    const unsigned short* __restrict__ wv, const float* __restrict__ bias3,  // [B,T,T]
    const float* __restrict__ gamma, const float* __restrict__ beta,
    const float* __restrict__ hscale, unsigned short* __restrict__ wo) {
  __shared__ unsigned short Ks[2][KT][LDK];    // key rows [t][dh], double-buffered
  __shared__ unsigned short Vs[2][DH_][LDK];   // V^T rows [dh][t], double-buffered
  __shared__ unsigned short Ps[4][16][LDK];    // per-wave P staging (C-layout -> A-layout)

  const int tid = threadIdx.x, lane = tid & 31, wid = tid >> 5;
  const int bh = blockIdx.y;
  const int b = bh >> 4, h = bh & 15;
  const int q0 = blockIdx.x * 64 + wid * 16;   // this wave's 16 query rows
  const int n16 = lane & 15, hi = lane >> 4;

  // async tile fetch: 8x16B per thread per key tile (K + V^T)
  auto issue_tile = [&](int buf, int kb) {
    const int kbase = kb * KT;
#pragma unroll
    for (int it = 0; it < 4; ++it) {
      int li = it * 128 + tid;
      int row = li >> 3;
      int c8 = (li & 7) << 3;
      async_copy16(&Ks[buf][row][c8], wk + ((size_t)bh * T_ + kbase + row) * DH_ + c8);
      async_copy16(&Vs[buf][row][c8], wv + ((size_t)bh * DH_ + row) * T_ + kbase + c8);
    }
  };

  // Q A-fragments (global bf16, K-contiguous), reused across all key tiles
  const unsigned short* qbase = wq + ((size_t)bh * T_ + q0) * DH_;
  v16bf aQ[2];
#pragma unroll
  for (int kk = 0; kk < 2; ++kk) aQ[kk] = frag_a(qbase + kk * 32, DH_, lane);

  float rowMax[8], rowSum[8];
  v8f oacc[4];
#pragma unroll
  for (int r = 0; r < 8; ++r) { rowMax[r] = -1e30f; rowSum[r] = 0.0f; }
#pragma unroll
  for (int j = 0; j < 4; ++j) oacc[j] = zero8();

  issue_tile(0, 0);
  for (int kb = 0; kb < NKT; ++kb) {
    const int buf = kb & 1;
    const int kbase = kb * KT;
    if (kb + 1 < NKT) {
      issue_tile(buf ^ 1, kb + 1);     // prefetch next tile into other buffer
      WAIT_ASYNC_LE(8);                // retire current tile's 8 copies
    } else {
      WAIT_ASYNC_LE(0);
    }
    __syncthreads();

    // S = Q K^T (Q pre-scaled), seeded with broadcast 3D bias
    v8f s[4];
#pragma unroll
    for (int j = 0; j < 4; ++j) {
      v8f sb;
#pragma unroll
      for (int r = 0; r < 8; ++r) {
        int gq = q0 + r + 8 * hi;
        sb[r] = bias3[((size_t)b * T_ + gq) * T_ + kbase + j * 16 + n16];
      }
#pragma unroll
      for (int kk = 0; kk < 2; ++kk) {
        v16bf bb = frag_b(&Ks[buf][j * 16][kk * 32], LDK, lane);
        sb = WMMA_BF16(aQ[kk], bb, sb);
      }
      s[j] = sb;
    }

    // online softmax (16-lane row reductions; rows sit on lanes n=0..15 per half)
#pragma unroll
    for (int r = 0; r < 8; ++r) {
      float mloc = fmaxf(fmaxf(s[0][r], s[1][r]), fmaxf(s[2][r], s[3][r]));
#pragma unroll
      for (int off = 8; off >= 1; off >>= 1) mloc = fmaxf(mloc, __shfl_xor(mloc, off, 32));
      float mnew = fmaxf(rowMax[r], mloc);
      float scl = __expf(rowMax[r] - mnew);
      rowMax[r] = mnew;
      float psum = 0.0f;
#pragma unroll
      for (int j = 0; j < 4; ++j) {
        float p = __expf(s[j][r] - mnew);
        s[j][r] = p;
        psum += p;
      }
#pragma unroll
      for (int off = 8; off >= 1; off >>= 1) psum += __shfl_xor(psum, off, 32);
      rowSum[r] = rowSum[r] * scl + psum;
#pragma unroll
      for (int j = 0; j < 4; ++j) oacc[j][r] *= scl;
    }

    // P: C-layout regs -> per-wave LDS (row-major [m][t]) -> A-layout frags
#pragma unroll
    for (int j = 0; j < 4; ++j)
#pragma unroll
      for (int r = 0; r < 8; ++r)
        Ps[wid][r + 8 * hi][j * 16 + n16] = f2bf(s[j][r]);
    asm volatile("s_wait_dscnt 0" ::: "memory");

    v16bf aP[2];
#pragma unroll
    for (int kk = 0; kk < 2; ++kk) aP[kk] = frag_a(&Ps[wid][0][kk * 32], LDK, lane);
#pragma unroll
    for (int j2 = 0; j2 < 4; ++j2)
#pragma unroll
      for (int kk = 0; kk < 2; ++kk) {
        v16bf bv = frag_b(&Vs[buf][j2 * 16][kk * 32], LDK, lane);
        oacc[j2] = WMMA_BF16(aP[kk], bv, oacc[j2]);
      }
    __syncthreads();
  }

  // epilogue: 1/l, per-head LayerNorm over DH=64, head scale, merged [B,T,D] bf16
  const float hs = hscale[h];
#pragma unroll
  for (int r = 0; r < 8; ++r) {
    float inv = 1.0f / rowSum[r];
    float o[4], sum = 0.0f, sq = 0.0f;
#pragma unroll
    for (int j = 0; j < 4; ++j) { o[j] = oacc[j][r] * inv; sum += o[j]; }
#pragma unroll
    for (int off = 8; off >= 1; off >>= 1) sum += __shfl_xor(sum, off, 32);
    float mean = sum * (1.0f / 64.0f);
#pragma unroll
    for (int j = 0; j < 4; ++j) { float d = o[j] - mean; sq += d * d; }
#pragma unroll
    for (int off = 8; off >= 1; off >>= 1) sq += __shfl_xor(sq, off, 32);
    float rstd = rsqrtf(sq * (1.0f / 64.0f) + EPS_);
    int t = q0 + r + 8 * hi;
#pragma unroll
    for (int j = 0; j < 4; ++j) {
      int dh = j * 16 + n16;
      float val = ((o[j] - mean) * rstd * gamma[dh] + beta[dh]) * hs;
      wo[((size_t)b * T_ + t) * D_ + h * DH_ + dh] = f2bf(val);
    }
  }
}

// ---------------------------------------------------------------- output GEMM (bf16 in, fp32 out)
__global__ __launch_bounds__(256) void out_gemm_kernel(
    const unsigned short* __restrict__ Xo,   // [B*T, D] bf16
    const unsigned short* __restrict__ Wo,   // [D, D] bf16
    const float* __restrict__ bo, float* __restrict__ out) {
  __shared__ unsigned short As[2][BM * LDT];
  __shared__ unsigned short Bs[2][BN * LDT];

  const int tid = threadIdx.x, lane = tid & 31, wid = tid >> 5;
  const int wm = wid >> 1, wn = wid & 1;
  const int rowBase = blockIdx.y * BM;
  const int colBase = blockIdx.x * BN;

  auto issue_tile = [&](int buf, int k0) {
#pragma unroll
    for (int it = 0; it < 2; ++it) {
      int li = it * 256 + tid;
      int row = li >> 2;
      int c8 = (li & 3) << 3;
      async_copy16(&As[buf][row * LDT + c8], Xo + (size_t)(rowBase + row) * D_ + k0 + c8);
      async_copy16(&Bs[buf][row * LDT + c8], Wo + (size_t)(colBase + row) * D_ + k0 + c8);
    }
  };

  v8f acc[2][4];
#pragma unroll
  for (int i = 0; i < 2; ++i)
#pragma unroll
    for (int j = 0; j < 4; ++j) acc[i][j] = zero8();

  issue_tile(0, 0);
  for (int kt = 0; kt < D_ / BK; ++kt) {
    const int buf = kt & 1;
    if (kt + 1 < D_ / BK) {
      issue_tile(buf ^ 1, (kt + 1) * BK);  // prefetch next K-slab
      WAIT_ASYNC_LE(4);                    // retire current slab's 4 copies
    } else {
      WAIT_ASYNC_LE(0);
    }
    __syncthreads();

    v16bf a0 = frag_a(&As[buf][(wm * 32) * LDT], LDT, lane);
    v16bf a1 = frag_a(&As[buf][(wm * 32 + 16) * LDT], LDT, lane);
#pragma unroll
    for (int nt = 0; nt < 4; ++nt) {
      v16bf bb = frag_b(&Bs[buf][(wn * 64 + nt * 16) * LDT], LDT, lane);
      acc[0][nt] = WMMA_BF16(a0, bb, acc[0][nt]);
      acc[1][nt] = WMMA_BF16(a1, bb, acc[1][nt]);
    }
    __syncthreads();
  }

  const int n16 = lane & 15, hi = lane >> 4;
#pragma unroll
  for (int mt = 0; mt < 2; ++mt)
#pragma unroll
    for (int nt = 0; nt < 4; ++nt)
#pragma unroll
      for (int r = 0; r < 8; ++r) {
        int gm = rowBase + wm * 32 + mt * 16 + r + 8 * hi;
        int gn = colBase + wn * 64 + nt * 16 + n16;
        out[(size_t)gm * D_ + gn] = acc[mt][nt][r] + bo[gn];
      }
}

// ----------------------------------------------------------------
extern "C" void kernel_launch(void* const* d_in, const int* in_sizes, int n_in,
                              void* d_out, int out_size, void* d_ws, size_t ws_size,
                              hipStream_t stream) {
  (void)in_sizes; (void)n_in; (void)out_size; (void)ws_size;
  const float* query  = (const float*)d_in[0];
  const float* key    = (const float*)d_in[1];
  const float* value  = (const float*)d_in[2];
  const float* bias3  = (const float*)d_in[3];
  const float* inw    = (const float*)d_in[4];
  const float* inb    = (const float*)d_in[5];
  const float* outw   = (const float*)d_in[6];
  const float* outb   = (const float*)d_in[7];
  const float* gamma  = (const float*)d_in[8];
  const float* beta   = (const float*)d_in[9];
  const float* hscale = (const float*)d_in[10];

  // workspace carve (bf16 elements): Q,K,V [B,H,T,DH] / [B,H,DH,T], O [B,T,D], weights bf16
  unsigned short* p = (unsigned short*)d_ws;
  unsigned short* wq = p;  p += (size_t)B_ * H_ * T_ * DH_;
  unsigned short* wk = p;  p += (size_t)B_ * H_ * T_ * DH_;
  unsigned short* wv = p;  p += (size_t)B_ * H_ * T_ * DH_;
  unsigned short* wo = p;  p += (size_t)B_ * T_ * D_;
  unsigned short* wIn = p; p += (size_t)3 * D_ * D_;
  unsigned short* wOut = p;

  cvt_bf16_kernel<<<dim3((3 * D_ * D_) / 256), 256, 0, stream>>>(inw, wIn, 3 * D_ * D_);
  cvt_bf16_kernel<<<dim3((D_ * D_) / 256), 256, 0, stream>>>(outw, wOut, D_ * D_);
  qkv_gemm_kernel<<<dim3(D_ / BN, (B_ * T_) / BM, 3), 256, 0, stream>>>(
      query, key, value, wIn, inb, wq, wk, wv);
  attn_kernel<<<dim3(T_ / KT, B_ * H_), 128, 0, stream>>>(
      wq, wk, wv, bias3, gamma, beta, hscale, wo);
  out_gemm_kernel<<<dim3(D_ / BN, (B_ * T_) / BM), 256, 0, stream>>>(
      wo, wOut, outb, (float*)d_out);
}
